// MaskedAttention_41506563948819
// MI455X (gfx1250) — matrix operand order repaired and checked by reading, not compiled, not run on hardware
//
#include <hip/hip_runtime.h>
#include <hip/hip_bf16.h>
#include <stdint.h>

// ---------------------------------------------------------------------------
// MaskedAttention for MI455X (gfx1250): bf16 WMMA 16x16x32 everywhere,
// fp32 accumulation, flash-attention with scores^T trick (no cross-lane
// shuffle between score-WMMA and PV-WMMA), V stored transposed, and a
// double-buffered GLOBAL_LOAD_ASYNC_TO_LDS pipeline for K/V (ASYNCcnt).
// ---------------------------------------------------------------------------

typedef __bf16 bf16;
typedef __attribute__((ext_vector_type(16))) __bf16 v16bf;
typedef __attribute__((ext_vector_type(8)))  float  v8f;
typedef int v4i32 __attribute__((vector_size(16)));

constexpr int kB  = 16;
constexpr int kE  = 256;
constexpr int kS  = 4096;   // 64*64
constexpr int kH  = 8;
constexpr int kD  = 32;
constexpr int kQ  = 100;
constexpr int kQp = 112;    // Q padded to 7 tiles of 16
constexpr int kLRow = 40;   // LDS row stride in elements (80B: conflict-free)

// Async global->LDS path (gfx1250). Guarded so the file compiles either way.
#if defined(__HIP_DEVICE_COMPILE__) && defined(__has_builtin)
#if __has_builtin(__builtin_amdgcn_global_load_async_to_lds_b128)
#define HAS_ASYNC_LDS 1
#endif
#endif
#ifndef HAS_ASYNC_LDS
#define HAS_ASYNC_LDS 0
#endif

struct U32x8 { uint4 a, b; };

__device__ __forceinline__ v16bf make16(uint4 x, uint4 y) {
  U32x8 t{x, y};
  return __builtin_bit_cast(v16bf, t);
}

__device__ __forceinline__ v8f wmma_bf16(v16bf a, v16bf b, v8f c) {
  // D = A(16x32) * B(32x16) + C, fp32 accumulate
  return __builtin_amdgcn_wmma_f32_16x16x32_bf16(false, a, false, b,
                                                 (short)0, c, false, false);
}

// half-wave swap (lanes L <-> L+16): ds_swizzle group-of-32, xor=0x10, and=0x1f
__device__ __forceinline__ float swz16(float x) {
  int i = __builtin_bit_cast(int, x);
  i = __builtin_amdgcn_ds_swizzle(i, 0x401f);
  return __builtin_bit_cast(float, i);
}

// broadcast value held in lane `idx` (0..15) to this lane
__device__ __forceinline__ float bcast_lane(float x, int idx) {
  int i = __builtin_amdgcn_ds_bpermute(idx << 2, __builtin_bit_cast(int, x));
  return __builtin_bit_cast(float, i);
}

__device__ __forceinline__ uint4 pack8bf(v8f c, float bias) {
  union { uint4 u; bf16 h[8]; } t;
#pragma unroll
  for (int r = 0; r < 8; ++r) t.h[r] = (bf16)(c[r] + bias);
  return t.u;
}

#if HAS_ASYNC_LDS
__device__ __forceinline__ void async_b128(const bf16* g, bf16* l) {
  __builtin_amdgcn_global_load_async_to_lds_b128(
      (__attribute__((address_space(1))) v4i32*)g,
      (__attribute__((address_space(3))) v4i32*)l, 0, 0);
}

// copy a 32-row tile: row i from gbase + i*gstride (32 elems) into LDS rows
// of kLRow elems. 4 instructions, each moving 32 lanes x 16B.
__device__ __forceinline__ void async_tile(const bf16* gbase, size_t gstride,
                                           bf16* lbase, int lane) {
  int row = lane >> 2, ch = (lane & 3) * 8;
#pragma unroll
  for (int i = 0; i < 4; ++i) {
    int r = row + i * 8;
    async_b128(gbase + (size_t)r * gstride + ch, lbase + r * kLRow + ch);
  }
}

template <int N>
__device__ __forceinline__ void wait_async() {
#if __has_builtin(__builtin_amdgcn_s_wait_asynccnt)
  __builtin_amdgcn_s_wait_asynccnt((short)N);
#else
  if constexpr (N == 0) asm volatile("s_wait_asynccnt 0x0" ::: "memory");
  else                  asm volatile("s_wait_asynccnt 0x8" ::: "memory");
#endif
}
#endif  // HAS_ASYNC_LDS

// ---------------------------------------------------------------------------
// Kernel 1: fp32 -> bf16 weight conversion (Wq|Wk|Wv|Wo packed into ws)
// ---------------------------------------------------------------------------
__global__ void wconv_kernel(const float* __restrict__ Wq, const float* __restrict__ Wk,
                             const float* __restrict__ Wv, const float* __restrict__ Wo,
                             bf16* __restrict__ out) {
  int i = blockIdx.x * blockDim.x + threadIdx.x;   // 0 .. 4*65536-1
  int which = i >> 16;
  int j = i & 0xFFFF;
  float v;
  if      (which == 0) v = Wq[j];
  else if (which == 1) v = Wk[j];
  else if (which == 2) v = Wv[j];
  else                 v = Wo[j];
  out[i] = (bf16)v;
}

// ---------------------------------------------------------------------------
// Kernel 2: transpose image_features (B,E,S) -> (B,S,E) bf16, and (+pos) bf16
// ---------------------------------------------------------------------------
__global__ void transpose_kernel(const float* __restrict__ imgf, const float* __restrict__ pos,
                                 bf16* __restrict__ img_bf, bf16* __restrict__ imgpos_bf) {
  __shared__ float t[32][33];
  int b = blockIdx.z, e0 = blockIdx.y * 32, s0 = blockIdx.x * 32;
  int tx = threadIdx.x, ty = threadIdx.y;          // 32 x 8
#pragma unroll
  for (int i = 0; i < 4; ++i) {
    int el = ty + i * 8;
    t[el][tx] = imgf[((size_t)(b * kE + e0 + el)) * kS + s0 + tx];
  }
  __syncthreads();
#pragma unroll
  for (int i = 0; i < 4; ++i) {
    int sl = ty + i * 8;
    size_t o = ((size_t)(b * kS + s0 + sl)) * kE + e0 + tx;
    float v = t[tx][sl];
    img_bf[o]    = (bf16)v;
    imgpos_bf[o] = (bf16)(v + pos[o]);
  }
}

// ---------------------------------------------------------------------------
// B-operand load for W (row-major [e_out][e] bf16): lane = column e_out,
// K = k0+half*8+0..7 and k0+half*8+16..23 per lane (same order as load_a).
// ---------------------------------------------------------------------------
__device__ __forceinline__ v16bf load_wb(const bf16* __restrict__ W, int n0, int k0,
                                         int l15, int half) {
  const bf16* p = W + (size_t)(n0 + l15) * kE + k0 + half * 8;
  return make16(*(const uint4*)p, *(const uint4*)(p + 16));
}

// A-operand load, bf16 row-major (row stride kE): lane<16 row=lane K=0..7,16..23
__device__ __forceinline__ v16bf load_a(const bf16* __restrict__ Arow, int k0,
                                        int l15, int half) {
  const bf16* p = Arow + (size_t)l15 * kE + k0 + half * 8;
  return make16(*(const uint4*)p, *(const uint4*)(p + 16));
}

// ---------------------------------------------------------------------------
// Kernel 3: K / V projection GEMM. A: (B*S,256) bf16, W: bf16, bias fp32.
//   TRANS_OUT=false -> out (B,H,S,D)  (K layout)
//   TRANS_OUT=true  -> out (B,H,D,S)  (V^T layout, packed b128 stores)
// Block: 256 thr = 8 waves; wave w owns columns [32w, 32w+32); M-tile = 16 rows.
// ---------------------------------------------------------------------------
template <bool TRANS_OUT>
__global__ void proj_kv_kernel(const bf16* __restrict__ A, const bf16* __restrict__ W,
                               const float* __restrict__ bias, bf16* __restrict__ out) {
  int lane = threadIdx.x & 31, wave = threadIdx.x >> 5;
  int l15 = lane & 15, half = (lane >> 4) & 1;
  int mt = blockIdx.x;                 // 0..4095
  int b  = mt >> 8;
  int m0 = (mt & 255) << 4;            // s-tile base
  const bf16* Arow = A + ((size_t)b * kS + m0) * kE;
  int n0 = wave * 32;

  // warm L2 for the next M-tile's activation rows (global_prefetch_b8)
  __builtin_prefetch(Arow + (size_t)16 * kE + (size_t)l15 * kE, 0, 1);

  v8f c0 = {}, c1 = {};
  for (int k0 = 0; k0 < kE; k0 += 32) {
    v16bf a  = load_a(Arow, k0, l15, half);
    v16bf b0 = load_wb(W, n0,      k0, l15, half);
    v16bf b1 = load_wb(W, n0 + 16, k0, l15, half);
    c0 = wmma_bf16(a, b0, c0);
    c1 = wmma_bf16(a, b1, c1);
  }
  int e0o = n0 + l15, e1o = n0 + 16 + l15;
  float bb0 = bias[e0o], bb1 = bias[e1o];
  int h0 = e0o >> 5, d0 = e0o & 31;
  int h1 = e1o >> 5, d1 = e1o & 31;
  if (TRANS_OUT) {
    int srow = m0 + half * 8;          // C rows are consecutive s
    *(uint4*)(out + (((size_t)(b * kH + h0) * kD + d0) * kS + srow)) = pack8bf(c0, bb0);
    *(uint4*)(out + (((size_t)(b * kH + h1) * kD + d1) * kS + srow)) = pack8bf(c1, bb1);
  } else {
#pragma unroll
    for (int r = 0; r < 8; ++r) {
      int s = m0 + half * 8 + r;
      out[((size_t)(b * kH + h0) * kS + s) * kD + d0] = (bf16)(c0[r] + bb0);
      out[((size_t)(b * kH + h1) * kS + s) * kD + d1] = (bf16)(c1[r] + bb1);
    }
  }
}

// ---------------------------------------------------------------------------
// Kernel 4: Q projection. A = (query+pos_query) fp32, rows padded to 112
// (pad rows contribute zero). out: Qp (B,H,112,D) bf16.
// ---------------------------------------------------------------------------
template <int I>
__device__ __forceinline__ void put4(v16bf& a, float4 f, float4 g, bool valid) {
  a[I + 0] = valid ? (bf16)(f.x + g.x) : (bf16)0.f;
  a[I + 1] = valid ? (bf16)(f.y + g.y) : (bf16)0.f;
  a[I + 2] = valid ? (bf16)(f.z + g.z) : (bf16)0.f;
  a[I + 3] = valid ? (bf16)(f.w + g.w) : (bf16)0.f;
}

__global__ void proj_q_kernel(const float* __restrict__ qf, const float* __restrict__ pq,
                              const bf16* __restrict__ W, const float* __restrict__ bias,
                              bf16* __restrict__ out) {
  int lane = threadIdx.x & 31, wave = threadIdx.x >> 5;
  int l15 = lane & 15, half = (lane >> 4) & 1;
  int b = blockIdx.x / 7, qt = blockIdx.x % 7;
  int qrow = qt * 16 + l15;
  bool valid = qrow < kQ;
  int qsafe = valid ? qrow : 0;
  const float* qb = qf + ((size_t)(b * kQ + qsafe)) * kE;
  const float* pb = pq + ((size_t)(b * kQ + qsafe)) * kE;
  int n0 = wave * 32;

  v8f c0 = {}, c1 = {};
  for (int k0 = 0; k0 < kE; k0 += 32) {
    int e = k0 + half * 8;
    const float4* q4 = (const float4*)(qb + e);
    const float4* p4 = (const float4*)(pb + e);
    v16bf a;
    put4<0>(a, q4[0], p4[0], valid);
    put4<4>(a, q4[1], p4[1], valid);
    put4<8>(a, q4[4], p4[4], valid);   // e+16..e+19
    put4<12>(a, q4[5], p4[5], valid);  // e+20..e+23
    v16bf b0 = load_wb(W, n0,      k0, l15, half);
    v16bf b1 = load_wb(W, n0 + 16, k0, l15, half);
    c0 = wmma_bf16(a, b0, c0);
    c1 = wmma_bf16(a, b1, c1);
  }
  int e0o = n0 + l15, e1o = n0 + 16 + l15;
  float bb0 = bias[e0o], bb1 = bias[e1o];
  int h0 = e0o >> 5, d0 = e0o & 31;
  int h1 = e1o >> 5, d1 = e1o & 31;
#pragma unroll
  for (int r = 0; r < 8; ++r) {
    int qg = qt * 16 + half * 8 + r;
    out[((size_t)(b * kH + h0) * kQp + qg) * kD + d0] = (bf16)(c0[r] + bb0);
    out[((size_t)(b * kH + h1) * kQp + qg) * kD + d1] = (bf16)(c1[r] + bb1);
  }
}

// ---------------------------------------------------------------------------
// Kernel 5: flash attention per (qtile, h, b). Block = 128 thr = 4 waves,
// wave w covers s in [1024w, 1024w+1024), 32 s per iteration.
// scores^T = K(16s x 32d) * Q^T(32d x 16q): lane owns one q column ->
// per-lane softmax state; C-fragment == A-operand layout for P*V.
// K/V staged in LDS through a 2-deep async pipeline (ASYNCcnt) when available.
// ---------------------------------------------------------------------------
__global__ void attn_kernel(const bf16* __restrict__ Qp, const bf16* __restrict__ Kp,
                            const bf16* __restrict__ Vt, const unsigned char* __restrict__ mask,
                            bf16* __restrict__ Oa) {
  constexpr float scale = 0.17677669529663687f;   // 1/sqrt(32)
  __shared__ float sm_[4][16];
  __shared__ float sl_[4][16];
  __shared__ float sacc[4][16][32];
#if HAS_ASYNC_LDS
  __shared__ bf16 kbuf[4][2][32 * kLRow];
  __shared__ bf16 vbuf[4][2][32 * kLRow];
#endif

  int lane = threadIdx.x & 31, wave = threadIdx.x >> 5;
  int l15 = lane & 15, half = (lane >> 4) & 1;
  int qt = blockIdx.x, h = blockIdx.y, b = blockIdx.z;
  int q0 = qt * 16;
  size_t bh = (size_t)(b * kH + h);

  // Q as B-operand: lane = column q, d = half*8+0..7 and half*8+16..23 per lane
  const bf16* pqv = Qp + (bh * kQp + q0 + l15) * kD + half * 8;
  v16bf qb = make16(*(const uint4*)pqv, *(const uint4*)(pqv + 16));

  int qrow = q0 + l15;
  int qc = qrow < kQ ? qrow : kQ - 1;              // clamp pad rows (unused)
  const unsigned char* mrow = mask + ((size_t)b * kQ + qc) * kS;

  const bf16* Kb = Kp + bh * (size_t)kS * kD;
  const bf16* Vb = Vt + bh * (size_t)kD * kS;

  float m = -1e30f, l = 0.f;
  v8f acc0 = {}, acc1 = {};

  int sbeg = wave * (kS / 4);
  constexpr int nIter = (kS / 4) / 32;             // 32

#if HAS_ASYNC_LDS
  async_tile(Kb + (size_t)sbeg * kD,        kD, &kbuf[wave][0][0], lane);
  async_tile(Vb + sbeg,                     kS, &vbuf[wave][0][0], lane);
  async_tile(Kb + (size_t)(sbeg + 32) * kD, kD, &kbuf[wave][1][0], lane);
  async_tile(Vb + sbeg + 32,                kS, &vbuf[wave][1][0], lane);
#endif

  for (int it = 0; it < nIter; ++it) {
    int s0 = sbeg + it * 32;
    v16bf ka0, ka1, vb0, vb1;
#if HAS_ASYNC_LDS
    if (it + 1 < nIter) wait_async<8>(); else wait_async<0>();
    bf16* kb = &kbuf[wave][it & 1][0];
    bf16* vb = &vbuf[wave][it & 1][0];
    const bf16* pk0 = kb + l15 * kLRow + half * 8;
    ka0 = make16(*(const uint4*)pk0, *(const uint4*)(pk0 + 16));
    const bf16* pk1 = kb + (l15 + 16) * kLRow + half * 8;
    ka1 = make16(*(const uint4*)pk1, *(const uint4*)(pk1 + 16));
    const bf16* pv0 = vb + l15 * kLRow + half * 8;
    vb0 = make16(*(const uint4*)pv0, *(const uint4*)(pv0 + 16));
    const bf16* pv1 = vb + (l15 + 16) * kLRow + half * 8;
    vb1 = make16(*(const uint4*)pv1, *(const uint4*)(pv1 + 16));
#else
    const bf16* pk0 = Kb + (size_t)(s0 + l15) * kD + half * 8;
    ka0 = make16(*(const uint4*)pk0, *(const uint4*)(pk0 + 16));
    const bf16* pk1 = pk0 + 16 * kD;
    ka1 = make16(*(const uint4*)pk1, *(const uint4*)(pk1 + 16));
    const bf16* pv0 = Vb + (size_t)l15 * kS + s0 + half * 8;
    vb0 = make16(*(const uint4*)pv0, *(const uint4*)(pv0 + 16));
    const bf16* pv1 = pv0 + (size_t)16 * kS;
    vb1 = make16(*(const uint4*)pv1, *(const uint4*)(pv1 + 16));
#endif

    v8f z = {};
    v8f sc0 = wmma_bf16(ka0, qb, z);               // scores^T, s in [s0, s0+16)
    v8f sc1 = wmma_bf16(ka1, qb, z);               // scores^T, s in [s0+16, s0+32)

    // --- mask + scale (contiguous 8 mask bytes per lane per tile) ---
    int smb = s0 + half * 8;
    uint2 mv0 = *(const uint2*)(mrow + smb);
    uint2 mv1 = *(const uint2*)(mrow + smb + 16);
#pragma unroll
    for (int r = 0; r < 8; ++r) {
      unsigned b0 = ((r < 4 ? mv0.x >> (8 * r) : mv0.y >> (8 * (r - 4)))) & 0xFFu;
      unsigned b1 = ((r < 4 ? mv1.x >> (8 * r) : mv1.y >> (8 * (r - 4)))) & 0xFFu;
      sc0[r] = b0 ? sc0[r] * scale : -1e9f;
      sc1[r] = b1 ? sc1[r] * scale : -1e9f;
    }

    // --- online softmax (per-lane q state, one half-wave merge) ---
    float bm = sc0[0];
#pragma unroll
    for (int r = 1; r < 8; ++r) bm = fmaxf(bm, sc0[r]);
#pragma unroll
    for (int r = 0; r < 8; ++r) bm = fmaxf(bm, sc1[r]);
    bm = fmaxf(bm, swz16(bm));                     // lanes L and L+16 share q
    float mn = fmaxf(m, bm);
    float alpha = __expf(m - mn);
    m = mn;
    float rs = 0.f;
#pragma unroll
    for (int r = 0; r < 8; ++r) { sc0[r] = __expf(sc0[r] - mn); rs += sc0[r]; }
#pragma unroll
    for (int r = 0; r < 8; ++r) { sc1[r] = __expf(sc1[r] - mn); rs += sc1[r]; }
    rs += swz16(rs);
    l = l * alpha + rs;

    // --- rescale accumulators: broadcast alpha(q=row) via ds_bpermute ---
#pragma unroll
    for (int r = 0; r < 8; ++r) {
      float br = bcast_lane(alpha, r + half * 8);
      acc0[r] *= br;
      acc1[r] *= br;
    }

    // --- P (scores^T C-fragment) is already the A-operand layout: pack ---
    v16bf pa;
#pragma unroll
    for (int r = 0; r < 8; ++r) pa[r] = (bf16)sc0[r];
#pragma unroll
    for (int r = 0; r < 8; ++r) pa[8 + r] = (bf16)sc1[r];

    acc0 = wmma_bf16(pa, vb0, acc0);               // out d in [0,16)
    acc1 = wmma_bf16(pa, vb1, acc1);               // out d in [16,32)

#if HAS_ASYNC_LDS
    if (it + 2 < nIter) {
      // ds reads of this stage completed before the WMMAs consumed them;
      // safe to refill the same stage now.
      asm volatile("" ::: "memory");
      int s2 = s0 + 64;
      async_tile(Kb + (size_t)s2 * kD, kD, &kbuf[wave][it & 1][0], lane);
      async_tile(Vb + s2,              kS, &vbuf[wave][it & 1][0], lane);
    }
#endif
  }

  // --- cross-wave combine in LDS ---
  if (half == 0) { sm_[wave][l15] = m; sl_[wave][l15] = l; }
#pragma unroll
  for (int r = 0; r < 8; ++r) {
    sacc[wave][r + half * 8][l15]      = acc0[r];
    sacc[wave][r + half * 8][l15 + 16] = acc1[r];
  }
  __syncthreads();

  int t = threadIdx.x;
  int qq = t >> 3;                // 0..15
  int db = (t & 7) * 4;           // 0,4,...,28
  float M = fmaxf(fmaxf(sm_[0][qq], sm_[1][qq]), fmaxf(sm_[2][qq], sm_[3][qq]));
  float e0 = __expf(sm_[0][qq] - M), e1 = __expf(sm_[1][qq] - M);
  float e2 = __expf(sm_[2][qq] - M), e3 = __expf(sm_[3][qq] - M);
  float L = e0 * sl_[0][qq] + e1 * sl_[1][qq] + e2 * sl_[2][qq] + e3 * sl_[3][qq];
  float inv = 1.0f / L;
  union { uint2 u; bf16 hh[4]; } o;
#pragma unroll
  for (int j = 0; j < 4; ++j) {
    float v = e0 * sacc[0][qq][db + j] + e1 * sacc[1][qq][db + j] +
              e2 * sacc[2][qq][db + j] + e3 * sacc[3][qq][db + j];
    o.hh[j] = (bf16)(v * inv);
  }
  *(uint2*)(Oa + ((size_t)b * kQp + q0 + qq) * kE + h * kD + db) = o.u;
}

// ---------------------------------------------------------------------------
// Kernel 6: output projection: Oa (B,112,256) bf16 @ Wo^T + bo -> fp32 out.
// ---------------------------------------------------------------------------
__global__ void proj_o_kernel(const bf16* __restrict__ A, const bf16* __restrict__ W,
                              const float* __restrict__ bias, float* __restrict__ out) {
  int lane = threadIdx.x & 31, wave = threadIdx.x >> 5;
  int l15 = lane & 15, half = (lane >> 4) & 1;
  int b = blockIdx.x / 7, qt = blockIdx.x % 7;
  const bf16* Arow = A + ((size_t)(b * kQp + qt * 16)) * kE;
  int n0 = wave * 32;

  v8f c0 = {}, c1 = {};
  for (int k0 = 0; k0 < kE; k0 += 32) {
    v16bf a  = load_a(Arow, k0, l15, half);
    v16bf b0 = load_wb(W, n0,      k0, l15, half);
    v16bf b1 = load_wb(W, n0 + 16, k0, l15, half);
    c0 = wmma_bf16(a, b0, c0);
    c1 = wmma_bf16(a, b1, c1);
  }
  int e0o = n0 + l15, e1o = n0 + 16 + l15;
  float bb0 = bias[e0o], bb1 = bias[e1o];
#pragma unroll
  for (int r = 0; r < 8; ++r) {
    int qg = qt * 16 + half * 8 + r;
    if (qg < kQ) {
      out[((size_t)(b * kQ + qg)) * kE + e0o] = c0[r] + bb0;
      out[((size_t)(b * kQ + qg)) * kE + e1o] = c1[r] + bb1;
    }
  }
}

// ---------------------------------------------------------------------------
extern "C" void kernel_launch(void* const* d_in, const int* in_sizes, int n_in,
                              void* d_out, int out_size, void* d_ws, size_t ws_size,
                              hipStream_t stream) {
  (void)in_sizes; (void)n_in; (void)out_size; (void)ws_size;
  const float* qf    = (const float*)d_in[0];
  const float* imgf  = (const float*)d_in[1];
  const unsigned char* mask = (const unsigned char*)d_in[2];  // jax bool = 1 byte
  const float* posq  = (const float*)d_in[3];
  const float* posi  = (const float*)d_in[4];
  const float* Wq = (const float*)d_in[5];
  const float* bq = (const float*)d_in[6];
  const float* Wk = (const float*)d_in[7];
  const float* bk = (const float*)d_in[8];
  const float* Wv = (const float*)d_in[9];
  const float* bv = (const float*)d_in[10];
  const float* Wo = (const float*)d_in[11];
  const float* bo = (const float*)d_in[12];
  float* out = (float*)d_out;

  // workspace carve (all chunk sizes are multiples of 128 KB -> aligned)
  char* p = (char*)d_ws;
  bf16* wbf       = (bf16*)p; p += (size_t)4 * kE * kE * sizeof(bf16);     // 512 KB
  bf16* img_bf    = (bf16*)p; p += (size_t)kB * kS * kE * sizeof(bf16);    // 32 MB
  bf16* imgpos_bf = (bf16*)p; p += (size_t)kB * kS * kE * sizeof(bf16);    // 32 MB
  bf16* Qp        = (bf16*)p; p += (size_t)kB * kH * kQp * kD * sizeof(bf16);
  bf16* Kp        = (bf16*)p; p += (size_t)kB * kH * kS * kD * sizeof(bf16);   // 32 MB
  bf16* Vt        = (bf16*)p; p += (size_t)kB * kH * kD * kS * sizeof(bf16);   // 32 MB
  bf16* Oa        = (bf16*)p; p += (size_t)kB * kQp * kE * sizeof(bf16);

  bf16* wq = wbf;
  bf16* wk = wbf + kE * kE;
  bf16* wv = wbf + 2 * kE * kE;
  bf16* wo = wbf + 3 * kE * kE;

  wconv_kernel<<<(4 * kE * kE) / 256, 256, 0, stream>>>(Wq, Wk, Wv, Wo, wbf);
  transpose_kernel<<<dim3(kS / 32, kE / 32, kB), dim3(32, 8), 0, stream>>>(
      imgf, posi, img_bf, imgpos_bf);
  proj_q_kernel<<<kB * 7, 256, 0, stream>>>(qf, posq, wq, bq, Qp);
  proj_kv_kernel<false><<<kB * (kS / 16), 256, 0, stream>>>(imgpos_bf, wk, bk, Kp);
  proj_kv_kernel<true ><<<kB * (kS / 16), 256, 0, stream>>>(img_bf,    wv, bv, Vt);
  attn_kernel<<<dim3(7, kH, kB), 128, 0, stream>>>(Qp, Kp, Vt, mask, Oa);
  proj_o_kernel<<<kB * 7, 256, 0, stream>>>(Oa, wo, bo, out);
}
